// NRL_6090263625919
// MI455X (gfx1250) — matrix operand order repaired and checked by reading, not compiled, not run on hardware
//
#include <hip/hip_runtime.h>
#include <hip/hip_bf16.h>

typedef __bf16 bf16;
typedef __attribute__((ext_vector_type(16))) __bf16 v16bf;
typedef __attribute__((ext_vector_type(8)))  __bf16 v8bf;
typedef __attribute__((ext_vector_type(8)))  float  v8f;

#define N_ROWS 16384
#define F_DIM  1024
#define H_DIM  256

// ---- workspace layout (bytes) ----
#define OFF_WET 0                               // bf16 WeT[H][F]   (512 KB)
#define OFF_WDT (F_DIM * H_DIM * 2)             // bf16 WdT[F][H]   (512 KB)
#define OFF_C   (2 * F_DIM * H_DIM * 2)         // f32 c[N]         (64 KB)
#define OFF_ACC (OFF_C + N_ROWS * 4)            // f32 acc[8]: 0=mse_sum 1=pair_total 2=pair_count

// ---------------------------------------------------------------------------
// Kernel 1: convert + transpose weights to bf16, zero accumulators.
// ---------------------------------------------------------------------------
__global__ void nrl_prep_kernel(const float* __restrict__ We,
                                const float* __restrict__ Wd,
                                bf16* __restrict__ WeT,
                                bf16* __restrict__ WdT,
                                float* __restrict__ acc) {
  int idx = blockIdx.x * blockDim.x + threadIdx.x;
  if (idx < F_DIM * H_DIM) {
    // We is [F,H] row-major -> WeT[h][f]
    int f = idx / H_DIM, h = idx % H_DIM;
    WeT[(size_t)h * F_DIM + f] = (bf16)We[(size_t)f * H_DIM + h];
    // Wd is [H,F] row-major -> WdT[f][h]
    int h2 = idx / F_DIM, f2 = idx % F_DIM;
    WdT[(size_t)f2 * H_DIM + h2] = (bf16)Wd[(size_t)h2 * F_DIM + f2];
  }
  if (idx < 8) acc[idx] = 0.0f;
}

// ---------------------------------------------------------------------------
// Kernel 2: fused encoder + critic + decoder + MSE partial reduction.
// One block = 256 threads = 8 waves, handles 16 rows of X.
// ---------------------------------------------------------------------------
__global__ __launch_bounds__(256) void nrl_fused_kernel(
    const float* __restrict__ X,   const bf16* __restrict__ WeT,
    const float* __restrict__ be,  const bf16* __restrict__ WdT,
    const float* __restrict__ bd,  const float* __restrict__ Wc,
    const float* __restrict__ bc,  float* __restrict__ cOut,
    float* __restrict__ acc) {
  __shared__ __align__(16) bf16  lX[16][F_DIM + 8];     // X tile, bf16
  __shared__ __align__(16) float lGf[16][H_DIM + 4];    // g tile, f32
  __shared__ __align__(16) bf16  lG[16][H_DIM];         // g tile, bf16
  __shared__ float red[256];

  const int tid     = threadIdx.x;
  const int lane    = tid & 31;
  const int wave    = tid >> 5;        // 0..7
  const int l15     = lane & 15;       // A-row / B-col / C-col index
  const int hi      = lane >> 4;       // 0 or 1 (K-half / M-half select)
  const int rowBase = blockIdx.x * 16;

  // Warm L2 with this wave's Wd panel for GEMM2 (global_prefetch_b8).
  __builtin_prefetch((const void*)(WdT + (size_t)(wave * 8 * 16 + l15) * H_DIM), 0, 1);

  // ---- stage 16 rows of X into LDS as bf16 (f32 -> bf16 on the fly) ----
  {
    int r  = tid >> 4;             // 16 threads per row
    int c0 = (tid & 15) * 64;      // 64 columns per thread
    const float4* src = (const float4*)(X + (size_t)(rowBase + r) * F_DIM + c0);
#pragma unroll
    for (int q = 0; q < 16; ++q) {
      float4 v = src[q];
      int c = c0 + q * 4;
      lX[r][c + 0] = (bf16)v.x; lX[r][c + 1] = (bf16)v.y;
      lX[r][c + 2] = (bf16)v.z; lX[r][c + 3] = (bf16)v.w;
    }
  }
  __syncthreads();

  // ---- GEMM1: g = X @ We + be. Wave owns H-tiles (2w) and (2w+1). ----
  v8f acc0, acc1;
  {
    float b0 = be[wave * 32 + l15];
    float b1 = be[wave * 32 + 16 + l15];
#pragma unroll
    for (int v = 0; v < 8; ++v) { acc0[v] = b0; acc1[v] = b1; }
  }
  for (int kc = 0; kc < 32; ++kc) {
    // A fragment: row = l15, K = kc*32 + hi*16 .. +15 (contiguous)
    int kOff = kc * 32 + hi * 16;
    v8bf alo = *(const v8bf*)&lX[l15][kOff];
    v8bf ahi = *(const v8bf*)&lX[l15][kOff + 8];
    v16bf a;
#pragma unroll
    for (int e = 0; e < 8; ++e) { a[e] = alo[e]; a[8 + e] = ahi[e]; }
    // B fragments: col = tile*16 + l15, K = kc*32 + hi*16 .. +15 (contiguous in WeT)
    int kb = kc * 32 + hi * 16;
    const v8bf* b0p = (const v8bf*)(WeT + (size_t)(wave * 32 + l15) * F_DIM + kb);
    const v8bf* b1p = (const v8bf*)(WeT + (size_t)(wave * 32 + 16 + l15) * F_DIM + kb);
    v8bf b0lo = b0p[0], b0hi = b0p[1];
    v8bf b1lo = b1p[0], b1hi = b1p[1];
    v16bf b0, b1;
#pragma unroll
    for (int e = 0; e < 8; ++e) {
      b0[e] = b0lo[e]; b0[8 + e] = b0hi[e];
      b1[e] = b1lo[e]; b1[8 + e] = b1hi[e];
    }
    acc0 = __builtin_amdgcn_wmma_f32_16x16x32_bf16(false, a, false, b0, (short)0, acc0, false, false);
    acc1 = __builtin_amdgcn_wmma_f32_16x16x32_bf16(false, a, false, b1, (short)0, acc1, false, false);
  }
  // write g (f32) to LDS:  C/D layout row M = v + 8*hi, col N = l15
#pragma unroll
  for (int v = 0; v < 8; ++v) {
    int r = v + hi * 8;
    lGf[r][wave * 32 + l15]      = acc0[v];
    lGf[r][wave * 32 + 16 + l15] = acc1[v];
  }
  __syncthreads();

  // ---- critic scores: c = g @ Wc + bc, fully parallel ----
  // 16 threads per row, each covers a 16-wide segment of H, then tree-finish.
  {
    int r    = tid >> 4;          // row 0..15
    int hseg = (tid & 15) * 16;   // segment base
    float partial = 0.0f;
#pragma unroll
    for (int q = 0; q < 16; ++q) partial += lGf[r][hseg + q] * Wc[hseg + q];
    red[tid] = partial;
  }
  __syncthreads();
  if (tid < 16) {
    float sum = bc[0];
#pragma unroll
    for (int q = 0; q < 16; ++q) sum += red[tid * 16 + q];
    cOut[rowBase + tid] = sum;
  }
  // convert g to bf16 for GEMM2 (independent of red[])
  for (int e = tid; e < 16 * H_DIM; e += 256) {
    int r = e >> 8, h = e & 255;
    lG[r][h] = (bf16)lGf[r][h];
  }
  __syncthreads();

  // ---- GEMM2: rec = g @ Wd + bd, fused with (rec - X)^2 reduction ----
  v8f racc[8];
#pragma unroll
  for (int t = 0; t < 8; ++t) {
    float b = bd[(wave * 8 + t) * 16 + l15];
#pragma unroll
    for (int v = 0; v < 8; ++v) racc[t][v] = b;
  }
  for (int kc = 0; kc < 8; ++kc) {
    int kOff = kc * 32 + hi * 16;
    v8bf alo = *(const v8bf*)&lG[l15][kOff];
    v8bf ahi = *(const v8bf*)&lG[l15][kOff + 8];
    v16bf a;
#pragma unroll
    for (int e = 0; e < 8; ++e) { a[e] = alo[e]; a[8 + e] = ahi[e]; }
    int kb = kc * 32 + hi * 16;
#pragma unroll
    for (int t = 0; t < 8; ++t) {
      int fcol = (wave * 8 + t) * 16 + l15;
      const v8bf* bp = (const v8bf*)(WdT + (size_t)fcol * H_DIM + kb);
      v8bf blo = bp[0], bhi = bp[1];
      v16bf b;
#pragma unroll
      for (int e = 0; e < 8; ++e) { b[e] = blo[e]; b[8 + e] = bhi[e]; }
      racc[t] = __builtin_amdgcn_wmma_f32_16x16x32_bf16(false, a, false, b, (short)0, racc[t], false, false);
    }
  }
  // epilogue: squared residual vs f32 X (L2-resident after staging pass)
  float sq = 0.0f;
#pragma unroll
  for (int t = 0; t < 8; ++t) {
    int fcol = (wave * 8 + t) * 16 + l15;
#pragma unroll
    for (int v = 0; v < 8; ++v) {
      int r = v + hi * 8;
      float xv = X[(size_t)(rowBase + r) * F_DIM + fcol];
      float d  = racc[t][v] - xv;
      sq += d * d;
    }
  }
  // block reduction -> global atomic
  __syncthreads();   // red[] reuse: critic consumers are done
  red[tid] = sq;
  __syncthreads();
  for (int sdd = 128; sdd > 0; sdd >>= 1) {
    if (tid < sdd) red[tid] += red[tid + sdd];
    __syncthreads();
  }
  if (tid == 0) atomicAdd(&acc[0], red[0]);
}

// ---------------------------------------------------------------------------
// Kernel 3: masked pairwise L1 sum + protected-row count (tiled N^2).
// ---------------------------------------------------------------------------
__global__ __launch_bounds__(256) void nrl_pairwise_kernel(
    const float* __restrict__ c, const int* __restrict__ s,
    const int* __restrict__ pPtr, float* __restrict__ acc) {
  __shared__ float cj[256];
  __shared__ float mr[256];
  __shared__ float red[256];
  const int tid = threadIdx.x;
  const int i   = blockIdx.x * 256 + tid;
  const int p   = pPtr[0];

  const float ci = c[i];
  const float mi = (s[i] == p) ? 1.0f : 0.0f;
  float local = 0.0f;

  for (int jt = 0; jt < N_ROWS / 256; ++jt) {
    int j = jt * 256 + tid;
    cj[tid] = c[j];
    mr[tid] = (s[j] != p) ? 1.0f : 0.0f;
    __syncthreads();
#pragma unroll 8
    for (int k = 0; k < 256; ++k) local += fabsf(ci - cj[k]) * mr[k];
    __syncthreads();
  }
  local *= mi;

  red[tid] = local;
  __syncthreads();
  for (int sdd = 128; sdd > 0; sdd >>= 1) {
    if (tid < sdd) red[tid] += red[tid + sdd];
    __syncthreads();
  }
  if (tid == 0) atomicAdd(&acc[1], red[0]);
  __syncthreads();

  red[tid] = mi;
  __syncthreads();
  for (int sdd = 128; sdd > 0; sdd >>= 1) {
    if (tid < sdd) red[tid] += red[tid + sdd];
    __syncthreads();
  }
  if (tid == 0) atomicAdd(&acc[2], red[0]);
}

// ---------------------------------------------------------------------------
// Kernel 4: finalize the four scalar outputs.
// ---------------------------------------------------------------------------
__global__ void nrl_finalize_kernel(const float* __restrict__ acc,
                                    float* __restrict__ out) {
  if (blockIdx.x == 0 && threadIdx.x == 0) {
    out[0] = acc[0] / (float)((long long)N_ROWS * (long long)F_DIM);  // mse_mean
    out[1] = (float)N_ROWS;                                          // rec.shape[0]
    float cnt = acc[2];
    out[2] = acc[1] / cnt;                                           // w_dist_mean
    out[3] = cnt;                                                    // w_dist_count
  }
}

// ---------------------------------------------------------------------------
extern "C" void kernel_launch(void* const* d_in, const int* in_sizes, int n_in,
                              void* d_out, int out_size, void* d_ws, size_t ws_size,
                              hipStream_t stream) {
  const float* X  = (const float*)d_in[0];
  const float* We = (const float*)d_in[1];
  const float* be = (const float*)d_in[2];
  const float* Wd = (const float*)d_in[3];
  const float* bd = (const float*)d_in[4];
  const float* Wc = (const float*)d_in[5];
  const float* bc = (const float*)d_in[6];
  const int*   s  = (const int*)d_in[7];
  const int*   p  = (const int*)d_in[8];

  char*  ws   = (char*)d_ws;
  bf16*  WeT  = (bf16*)(ws + OFF_WET);
  bf16*  WdT  = (bf16*)(ws + OFF_WDT);
  float* cbuf = (float*)(ws + OFF_C);
  float* accd = (float*)(ws + OFF_ACC);
  float* out  = (float*)d_out;

  nrl_prep_kernel<<<(F_DIM * H_DIM + 255) / 256, 256, 0, stream>>>(We, Wd, WeT, WdT, accd);
  nrl_fused_kernel<<<N_ROWS / 16, 256, 0, stream>>>(X, WeT, be, WdT, bd, Wc, bc, cbuf, accd);
  nrl_pairwise_kernel<<<N_ROWS / 256, 256, 0, stream>>>(cbuf, s, p, accd);
  nrl_finalize_kernel<<<1, 64, 0, stream>>>(accd, out);
}
